// PolynomialDemandModel_31585189495219
// MI455X (gfx1250) — matrix operand order repaired and checked by reading, not compiled, not run on hardware
//
#include <hip/hip_runtime.h>

// ---------------- problem constants ----------------
#define BATCH          65536
#define IN_DIM         64
#define OUT_DIM        64
#define N_PRICES       16
#define NPAIRS         2016
#define KLIN           64
#define KQUAD          48
#define KTOT_REAL      2128                  // 64 + 48 + 2016
#define KCHUNK         32
#define NCHUNKS        67                    // ceil(2128/32)
#define KTOT_PAD       (NCHUNKS * KCHUNK)    // 2144
#define NTILES         4                     // 64 outputs / 16

// ---------------- launch geometry ----------------
#define BLOCK_THREADS  256                   // 8 wave32
#define ROWS_PER_WAVE  32
#define ROWS_PER_BLOCK 256

// ---------------- LDS layout ----------------
#define XROW_STRIDE    67                    // 64 data + 1.0 + 0.0 + pad (odd stride: no bank conflicts)
#define XS_BYTES       (ROWS_PER_BLOCK * XROW_STRIDE * 4)   // 68608
#define DESC_BYTES     (KTOT_PAD * 4)                       // 8576
#define BCHUNK_BYTES   (NTILES * 32 * 16 * 2)               // 4096 (one K-chunk of B frags)
#define BBUF_OFF       (XS_BYTES + DESC_BYTES)              // 77184
#define LDS_BYTES      (BBUF_OFF + 2 * BCHUNK_BYTES)        // 85376

typedef __attribute__((ext_vector_type(16))) _Float16 v16h;
typedef __attribute__((ext_vector_type(8)))  float    v8f;

// =====================================================================
// Prep kernel 1: feature descriptor table.
// Every feature k is x[j]*x[k2]: linear -> (k, 64 /*=1.0*/),
// quadratic -> (16+i, 16+i), pair p -> (j, k2) in triu(j<k2) order, pad -> (65,65 /*=0.0*/)
// =====================================================================
__global__ __launch_bounds__(256) void build_desc_kernel(unsigned int* __restrict__ desc) {
  int k = blockIdx.x * blockDim.x + threadIdx.x;
  if (k >= KTOT_PAD) return;
  unsigned int j, k2;
  if (k < KLIN) {
    j = (unsigned int)k; k2 = 64u;
  } else if (k < KLIN + KQUAD) {
    j = (unsigned int)(k - KLIN + N_PRICES); k2 = j;
  } else if (k < KTOT_REAL) {
    int rem = k - (KLIN + KQUAD);
    int jj = 0;
    while (rem >= (IN_DIM - 1 - jj)) { rem -= (IN_DIM - 1 - jj); ++jj; }
    j = (unsigned int)jj; k2 = (unsigned int)(jj + 1 + rem);
  } else {
    j = 65u; k2 = 65u;                 // zero column -> padded K contributes 0
  }
  desc[k] = j | (k2 << 8);
}

// =====================================================================
// Prep kernel 2: pack concat(linear | quadratic | interaction) -> f16,
// laid out directly in WMMA B-fragment order:
//   element (chunk c, n-tile t, lane, h) holds W[k = c*32 + (lane>>4)*16 + h][n = t*16 + (lane&15)]
// so the hot loop reads one 32B contiguous span per lane per fragment.
// =====================================================================
__global__ __launch_bounds__(256) void build_bfrag_kernel(const float* __restrict__ linear,
                                                          const float* __restrict__ quadratic,
                                                          const float* __restrict__ interaction,
                                                          _Float16* __restrict__ bf) {
  int idx = blockIdx.x * blockDim.x + threadIdx.x;
  if (idx >= NCHUNKS * NTILES * 32 * 16) return;
  int h    = idx & 15;
  int lane = (idx >> 4) & 31;
  int t    = (idx >> 9) & 3;
  int c    = idx >> 11;
  int n    = t * 16 + (lane & 15);
  int k    = c * KCHUNK + (lane >> 4) * 16 + h;
  float w;
  if (k < KLIN)                 w = linear[n * IN_DIM + k];
  else if (k < KLIN + KQUAD)    w = quadratic[n * KQUAD + (k - KLIN)];
  else if (k < KTOT_REAL)       w = interaction[n * NPAIRS + (k - KLIN - KQUAD)];
  else                          w = 0.0f;
  bf[idx] = (_Float16)w;
}

// =====================================================================
// Main kernel: fused feature-build + GEMM via v_wmma_f32_16x16x32_f16.
// Wave tile: M=32 rows x N=64 outputs. B frags double-buffered in LDS via
// global_load_async_to_lds_b128 and shared by all 8 waves.
// =====================================================================
__global__ __launch_bounds__(BLOCK_THREADS)
void poly_wmma_kernel(const float* __restrict__ x,
                      const float* __restrict__ bias,
                      const unsigned int* __restrict__ desc_g,
                      const _Float16* __restrict__ bfrag_g,
                      float* __restrict__ out)
{
  extern __shared__ char smem[];
  float*        xs    = (float*)smem;
  unsigned int* descs = (unsigned int*)(smem + XS_BYTES);

  const int tid  = threadIdx.x;
  const int lane = tid & 31;
  const int wave = tid >> 5;
  const int m    = lane & 15;   // row-in-tile for A / column for B,C,D
  const int g    = lane >> 4;   // lane group
  const long long block_row0 = (long long)blockIdx.x * ROWS_PER_BLOCK;

  // ---- stage x tile (256 x 64) into LDS, append virtual columns 1.0 / 0.0
  for (int i = tid; i < ROWS_PER_BLOCK * IN_DIM; i += BLOCK_THREADS) {
    int r = i >> 6, cc = i & 63;
    xs[r * XROW_STRIDE + cc] = x[(block_row0 + r) * IN_DIM + cc];
  }
  for (int r = tid; r < ROWS_PER_BLOCK; r += BLOCK_THREADS) {
    xs[r * XROW_STRIDE + 64] = 1.0f;
    xs[r * XROW_STRIDE + 65] = 0.0f;
    xs[r * XROW_STRIDE + 66] = 0.0f;
  }
  for (int i = tid; i < KTOT_PAD; i += BLOCK_THREADS) descs[i] = desc_g[i];

  // ---- async prefetch of chunk 0 B fragments into LDS buffer 0 (CDNA5 async path)
  {
    unsigned int ldsoff = (unsigned int)BBUF_OFF + (unsigned int)tid * 16u;
    const char*  gsrc   = (const char*)bfrag_g + (size_t)tid * 16u;
    asm volatile("global_load_async_to_lds_b128 %0, %1, off"
                 :: "v"(ldsoff), "v"(gsrc) : "memory");
  }
  asm volatile("s_wait_asynccnt 0" ::: "memory");
  __syncthreads();

  // ---- accumulators initialized with bias (bias depends only on column n)
  float bn[NTILES];
#pragma unroll
  for (int t = 0; t < NTILES; ++t) bn[t] = bias[t * 16 + m];
  v8f acc[2][NTILES];
#pragma unroll
  for (int mt = 0; mt < 2; ++mt)
#pragma unroll
    for (int t = 0; t < NTILES; ++t)
#pragma unroll
      for (int r = 0; r < 8; ++r) acc[mt][t][r] = bn[t];

  const int lrow0 = wave * ROWS_PER_WAVE;
  const float* xr0 = xs + (lrow0 + m) * XROW_STRIDE;        // rows 0..15 of this wave tile
  const float* xr1 = xs + (lrow0 + 16 + m) * XROW_STRIDE;   // rows 16..31

  for (int c = 0; c < NCHUNKS; ++c) {
    // prefetch next chunk into the other LDS buffer
    if (c + 1 < NCHUNKS) {
      unsigned int ldsoff = (unsigned int)(BBUF_OFF + ((c + 1) & 1) * BCHUNK_BYTES)
                          + (unsigned int)tid * 16u;
      const char*  gsrc   = (const char*)bfrag_g
                          + (size_t)(c + 1) * BCHUNK_BYTES + (size_t)tid * 16u;
      asm volatile("global_load_async_to_lds_b128 %0, %1, off"
                   :: "v"(ldsoff), "v"(gsrc) : "memory");
    }

    // B fragments for this chunk (32B contiguous per lane, from LDS)
    const char* bb = smem + BBUF_OFF + (c & 1) * BCHUNK_BYTES;
    v16h bfr[NTILES];
#pragma unroll
    for (int t = 0; t < NTILES; ++t)
      bfr[t] = *(const v16h*)(bb + (t * 32 + lane) * 32);

    // feature descriptors for this chunk (uniform across the 16 lanes of a group -> LDS broadcast)
    const int kbase = c * KCHUNK;
    unsigned int d0[8], d1[8];
#pragma unroll
    for (int v = 0; v < 8; ++v) {
      int kl = (g << 4) + (v << 1);  // contiguous K layout: k = 16*g + s (matches B frags)
      d0[v] = descs[kbase + kl];
      d1[v] = descs[kbase + kl + 1];
    }

    // build both A fragments: feature = x[j] * x[k2] (covers linear/quad/pair/pad uniformly)
    union { v16h v; _Float16 h[16]; } a0, a1;
#pragma unroll
    for (int v = 0; v < 8; ++v) {
      float f;
      f = xr0[d0[v] & 0xFFu] * xr0[(d0[v] >> 8) & 0xFFu]; a0.h[2 * v]     = (_Float16)f;
      f = xr0[d1[v] & 0xFFu] * xr0[(d1[v] >> 8) & 0xFFu]; a0.h[2 * v + 1] = (_Float16)f;
      f = xr1[d0[v] & 0xFFu] * xr1[(d0[v] >> 8) & 0xFFu]; a1.h[2 * v]     = (_Float16)f;
      f = xr1[d1[v] & 0xFFu] * xr1[(d1[v] >> 8) & 0xFFu]; a1.h[2 * v + 1] = (_Float16)f;
    }

    // 8 WMMAs: 2 row sub-tiles x 4 output tiles, f32 accumulation
#pragma unroll
    for (int t = 0; t < NTILES; ++t) {
      acc[0][t] = __builtin_amdgcn_wmma_f32_16x16x32_f16(false, a0.v, false, bfr[t],
                                                         (short)0, acc[0][t], false, false);
      acc[1][t] = __builtin_amdgcn_wmma_f32_16x16x32_f16(false, a1.v, false, bfr[t],
                                                         (short)0, acc[1][t], false, false);
    }

    // drain our async prefetch, then block-wide handshake before buffer swap
    asm volatile("s_wait_asynccnt 0" ::: "memory");
    __syncthreads();
  }

  // ---- epilogue: C/D layout -> row = mt*16 + g*8 + r, col = t*16 + m
#pragma unroll
  for (int mt = 0; mt < 2; ++mt) {
#pragma unroll
    for (int r = 0; r < 8; ++r) {
      long long row = block_row0 + lrow0 + mt * 16 + g * 8 + r;
      float* op = out + row * OUT_DIM + m;
#pragma unroll
      for (int t = 0; t < NTILES; ++t)
        op[t * 16] = acc[mt][t][r];
    }
  }
}

// =====================================================================
// Host entry
// =====================================================================
extern "C" void kernel_launch(void* const* d_in, const int* in_sizes, int n_in,
                              void* d_out, int out_size, void* d_ws, size_t ws_size,
                              hipStream_t stream) {
  const float* x           = (const float*)d_in[0];
  const float* linear      = (const float*)d_in[1];
  const float* quadratic   = (const float*)d_in[2];
  const float* interaction = (const float*)d_in[3];
  const float* bias        = (const float*)d_in[4];
  float* out = (float*)d_out;

  // workspace: [0, 8576) desc table, [16384, 16384+274432) f16 B fragments
  unsigned int* desc  = (unsigned int*)d_ws;
  _Float16*     bfrag = (_Float16*)((char*)d_ws + 16384);

  build_desc_kernel<<<(KTOT_PAD + 255) / 256, 256, 0, stream>>>(desc);

  int nbf = NCHUNKS * NTILES * 32 * 16;
  build_bfrag_kernel<<<(nbf + 255) / 256, 256, 0, stream>>>(linear, quadratic, interaction, bfrag);

  poly_wmma_kernel<<<BATCH / ROWS_PER_BLOCK, BLOCK_THREADS, LDS_BYTES, stream>>>(
      x, bias, desc, bfrag, out);
}
